// Transformer_55121610276942
// MI455X (gfx1250) — compile-verified, matrix operand order checked by reading, not run on hardware
//
#include <hip/hip_runtime.h>
#include <hip/hip_bf16.h>

// ---------------- model constants (match reference) ----------------
constexpr int Mm = 3, Bb = 128, Nn = 32, Dd = 512;
constexpr int Hh = 8, DHh = 64;
constexpr int Ff = 2048, SFf = 3 * 2048;
constexpr int DEPTHd = 2, Ee = 1024, OUTo = 64;
constexpr int Rr = Bb * Nn;                 // rows per modality = 4096
constexpr long long S1 = (long long)Rr * Dd;     // 2,097,152
constexpr long long SS = (long long)Mm * S1;     // 6,291,456
constexpr int KK = Mm * Nn * Dd;            // 49152

typedef __attribute__((ext_vector_type(16))) __bf16 v16bf;
typedef __attribute__((ext_vector_type(8)))  float  v8f;

// ---------------- WMMA bf16 GEMM: out = act(A*W + bias + res) ----------------
// A [rows,K] f32, W [K,N] f32, per-z strides. 64x64 tile / block, 8 waves,
// each wave -> 16x32 strip via two v_wmma_f32_16x16x32_bf16 accumulators.
// A tile staged raw-f32 into LDS with GLOBAL_LOAD_ASYNC_TO_LDS_B128 (ASYNCcnt
// DMA, no VGPR round trip); f32->bf16 happens at fragment build so the cvts
// co-execute with the XDL WMMAs. W tile is transposed into LDS as bf16.
__global__ void gemm_wmma(const float* __restrict__ A, long long aStr,
                          const float* __restrict__ W, long long wStr,
                          const float* __restrict__ bias, long long bStr,
                          const float* __restrict__ res, long long rStr,
                          float* __restrict__ out, long long oStr,
                          int K, int N, int doRelu)
{
    __shared__ float  Asf[64][36];  // [row][k] f32, async-DMA target (stride 144B, 16B aligned)
    __shared__ __bf16 Bs[64][40];   // [col][k] bf16 (transposed W tile, 8 pad)

    const int z = blockIdx.z;
    const float* Ab = A + (long long)z * aStr;
    const float* Wb = W + (long long)z * wStr;
    const float* Bi = bias ? bias + (long long)z * bStr : nullptr;
    const float* Rb = res  ? res  + (long long)z * rStr : nullptr;
    float*       Ob = out + (long long)z * oStr;

    const int rowBase = blockIdx.y * 64;
    const int colBase = blockIdx.x * 64;
    const int t    = threadIdx.x;
    const int lane = t & 31;
    const int wv   = t >> 5;
    const int wr   = wv & 3;        // wave row group (0..3)
    const int wc   = wv >> 2;       // wave col group (0..1)
    const int hf   = lane >> 4;     // lane half
    const int l16  = lane & 15;

    // per-thread staging coordinates
    const int ar8  = t >> 2, aseg = t & 3;       // A: row 0..63, 8-float segment 0..3
    const int bkk  = t >> 3, bseg = t & 7;       // B: k-row 0..31, 8-col segment 0..7
    const unsigned asBase = (unsigned)(size_t)&Asf[0][0];
    const unsigned ldsA   = asBase + (unsigned)(ar8 * 36 + aseg * 8) * 4u;

    v8f acc0 = {}, acc1 = {};

    for (int k0 = 0; k0 < K; k0 += 32) {
        {   // stage A tile 64x32 f32 via async DMA to LDS (2 x B128 per thread)
            const float* gp = Ab + (long long)(rowBase + ar8) * K + k0 + aseg * 8;
            unsigned long long ga = (unsigned long long)(size_t)gp;
            asm volatile("global_load_async_to_lds_b128 %0, %1, off"
                         :: "v"(ldsA), "v"(ga) : "memory");
            asm volatile("global_load_async_to_lds_b128 %0, %1, off"
                         :: "v"(ldsA + 16u), "v"(ga + 16ull) : "memory");
        }
        {   // stage W tile 32x64 transposed -> Bs[col][k] (bf16)
            const float4* p = (const float4*)(Wb + (long long)(k0 + bkk) * N + colBase + bseg * 8);
            float4 f0 = p[0], f1 = p[1];
            const int c = bseg * 8;
            Bs[c + 0][bkk] = (__bf16)f0.x; Bs[c + 1][bkk] = (__bf16)f0.y;
            Bs[c + 2][bkk] = (__bf16)f0.z; Bs[c + 3][bkk] = (__bf16)f0.w;
            Bs[c + 4][bkk] = (__bf16)f1.x; Bs[c + 5][bkk] = (__bf16)f1.y;
            Bs[c + 6][bkk] = (__bf16)f1.z; Bs[c + 7][bkk] = (__bf16)f1.w;
        }
        if (k0 + 32 < K) {  // prefetch next weight tile (global_prefetch_b8)
            __builtin_prefetch(Wb + (long long)(k0 + 32 + bkk) * N + colBase + bseg * 8, 0, 1);
        }
        asm volatile("s_wait_asynccnt 0x0" ::: "memory");
        __syncthreads();

        // ISA 7.12.2 16-bit A layout: lane half 0 -> K {0..7,16..23}, half 1 -> +8
        v16bf a, b0, b1;
        const int arow = wr * 16 + l16;
        const int c0 = wc * 32 + l16;
        const int c1 = c0 + 16;
#pragma unroll
        for (int vv = 0; vv < 8; ++vv) {
            const int kb = (vv < 4) ? (vv * 2 + hf * 8) : (16 + (vv - 4) * 2 + hf * 8);
            a [2 * vv] = (__bf16)Asf[arow][kb]; a [2 * vv + 1] = (__bf16)Asf[arow][kb + 1];
            b0[2 * vv] = Bs[c0][kb];            b0[2 * vv + 1] = Bs[c0][kb + 1];
            b1[2 * vv] = Bs[c1][kb];            b1[2 * vv + 1] = Bs[c1][kb + 1];
        }
        acc0 = __builtin_amdgcn_wmma_f32_16x16x32_bf16(false, a, false, b0, (short)0, acc0, false, false);
        acc1 = __builtin_amdgcn_wmma_f32_16x16x32_bf16(false, a, false, b1, (short)0, acc1, false, false);
        __syncthreads();
    }

    // epilogue: C/D layout: col = lane%16, row = vgpr + 8*(lane/16)
    const int gr0  = rowBase + wr * 16 + hf * 8;
    const int gc0  = colBase + wc * 32 + l16;
    const int gc1  = gc0 + 16;
    const float b0v = Bi ? Bi[gc0] : 0.f;
    const float b1v = Bi ? Bi[gc1] : 0.f;
    float o0[8], o1[8];
#pragma unroll
    for (int r = 0; r < 8; ++r) { o0[r] = acc0[r] + b0v; o1[r] = acc1[r] + b1v; }
    if (Rb) {
#pragma unroll
        for (int r = 0; r < 8; ++r) {
            o0[r] += Rb[(long long)(gr0 + r) * N + gc0];
            o1[r] += Rb[(long long)(gr0 + r) * N + gc1];
        }
    }
    if (doRelu) {
#pragma unroll
        for (int r = 0; r < 8; ++r) { o0[r] = fmaxf(o0[r], 0.f); o1[r] = fmaxf(o1[r], 0.f); }
    }
#pragma unroll
    for (int r = 0; r < 8; ++r) {
        Ob[(long long)(gr0 + r) * N + gc0] = o0[r];
        Ob[(long long)(gr0 + r) * N + gc1] = o1[r];
    }
}

// ---------------- row LayerNorm (rowLen elems, per-group gamma/beta) --------
__global__ void ln_kernel(const float* __restrict__ in, float* __restrict__ out,
                          const float* __restrict__ g, const float* __restrict__ b,
                          int rowLen, int rowsPerGroup)
{
    __shared__ float s1[256], s2[256];
    const long long row = blockIdx.x;
    const float* x = in + row * rowLen;
    float*       y = out + row * rowLen;
    const int grp = (int)(row / rowsPerGroup);
    const float* gg = g + (long long)grp * rowLen;
    const float* bb = b + (long long)grp * rowLen;
    const int t = threadIdx.x;
    float sum = 0.f, sq = 0.f;
    for (int i = t; i < rowLen; i += 256) { float v = x[i]; sum += v; sq += v * v; }
    s1[t] = sum; s2[t] = sq; __syncthreads();
    for (int o = 128; o > 0; o >>= 1) {
        if (t < o) { s1[t] += s1[t + o]; s2[t] += s2[t + o]; }
        __syncthreads();
    }
    const float mean = s1[0] / rowLen;
    const float var  = s2[0] / rowLen - mean * mean;
    const float inv  = rsqrtf(var + 1e-5f);
    for (int i = t; i < rowLen; i += 256) y[i] = (x[i] - mean) * inv * gg[i] + bb[i];
}

// ---------------- flatten [M,B,N,D]->[B,K] + LayerNorm over K ----------------
__global__ void final_ln_kernel(const float* __restrict__ x, float* __restrict__ out,
                                const float* __restrict__ g, const float* __restrict__ b)
{
    __shared__ float s1[256], s2[256];
    const int bb = blockIdx.x, t = threadIdx.x;
    const int ND = Nn * Dd;
    float sum = 0.f, sq = 0.f;
    for (int k = t; k < KK; k += 256) {
        const int m = k / ND, r = k % ND;
        const float v = x[(long long)m * Bb * ND + (long long)bb * ND + r];
        sum += v; sq += v * v;
    }
    s1[t] = sum; s2[t] = sq; __syncthreads();
    for (int o = 128; o > 0; o >>= 1) {
        if (t < o) { s1[t] += s1[t + o]; s2[t] += s2[t + o]; }
        __syncthreads();
    }
    const float mean = s1[0] / KK;
    const float var  = s2[0] / KK - mean * mean;
    const float inv  = rsqrtf(var + 1e-5f);
    for (int k = t; k < KK; k += 256) {
        const int m = k / ND, r = k % ND;
        const float v = x[(long long)m * Bb * ND + (long long)bb * ND + r];
        out[(long long)bb * KK + k] = (v - mean) * inv * g[k] + b[k];
    }
}

// ---------------- small-tile attention: softmax(relu(q)relu(k)^T*s)relu(v) ---
__global__ void attn_kernel(const float* __restrict__ q, const float* __restrict__ k,
                            const float* __restrict__ v, float* __restrict__ out)
{
    __shared__ float qs[32][64], ks[32][64], vs[32][64];
    __shared__ float sc[32][33];
    const int idx = blockIdx.x;
    const int h = idx % Hh;
    const int b = (idx / Hh) % Bb;
    const int m = idx / (Hh * Bb);
    const long long base = ((long long)(m * Bb + b)) * Nn * Dd + h * DHh;
    const int t = threadIdx.x;
    for (int e = t; e < 32 * 64; e += 256) {
        const int n = e >> 6, dd = e & 63;
        const long long gidx = base + (long long)n * Dd + dd;
        qs[n][dd] = fmaxf(q[gidx], 0.f);
        ks[n][dd] = fmaxf(k[gidx], 0.f);
        vs[n][dd] = fmaxf(v[gidx], 0.f);
    }
    __syncthreads();
    for (int e = t; e < 32 * 32; e += 256) {
        const int i = e >> 5, j = e & 31;
        float s = 0.f;
#pragma unroll 8
        for (int dd = 0; dd < 64; ++dd) s += qs[i][dd] * ks[j][dd];
        sc[i][j] = s * 0.125f;   // DH^-0.5
    }
    __syncthreads();
    if (t < 32) {
        float mx = -1e30f;
        for (int j = 0; j < 32; ++j) mx = fmaxf(mx, sc[t][j]);
        float sum = 0.f;
        for (int j = 0; j < 32; ++j) { const float e2 = __expf(sc[t][j] - mx); sc[t][j] = e2; sum += e2; }
        const float inv = 1.f / sum;
        for (int j = 0; j < 32; ++j) sc[t][j] *= inv;
    }
    __syncthreads();
    for (int e = t; e < 32 * 64; e += 256) {
        const int n = e >> 6, dd = e & 63;
        float s = 0.f;
#pragma unroll 8
        for (int j = 0; j < 32; ++j) s += sc[n][j] * vs[j][dd];
        out[base + (long long)n * Dd + dd] = s;
    }
}

// ---------------- cross-channel mix: s_i = w_i0*h_j0 + w_i1*h_j1 -------------
__global__ void mix_kernel(const float* __restrict__ h, const float* __restrict__ cw,
                           float* __restrict__ out)
{
    const long long idx = (long long)blockIdx.x * 256 + threadIdx.x;
    if (idx >= SS) return;
    const int m = (int)(idx / S1);
    const long long r = idx % S1;
    const int j0 = (m == 0) ? 1 : 0;
    const int j1 = (m == 2) ? 1 : 2;
    const float w0 = cw[m * Mm + 0], w1 = cw[m * Mm + 1];
    out[idx] = w0 * h[(long long)j0 * S1 + r] + w1 * h[(long long)j1 * S1 + r];
}

// ---------------- host orchestration ----------------------------------------
static inline void launch_gemm(hipStream_t st,
                               const float* A, long long aStr,
                               const float* W, long long wStr,
                               const float* bi, long long bStr,
                               const float* rs, long long rStr,
                               float* o, long long oStr,
                               int rows, int K, int N, int nz, int relu)
{
    dim3 g(N / 64, rows / 64, nz), b(256, 1, 1);
    gemm_wmma<<<g, b, 0, st>>>(A, aStr, W, wStr, bi, bStr, rs, rStr, o, oStr, K, N, relu);
}

extern "C" void kernel_launch(void* const* d_in, const int* in_sizes, int n_in,
                              void* d_out, int out_size, void* d_ws, size_t ws_size,
                              hipStream_t stream)
{
    typedef long long ll;
    const float* x_in = (const float*)d_in[0];
    const float* qW  = (const float*)d_in[1];
    const float* kW  = (const float*)d_in[2];
    const float* vW  = (const float*)d_in[3];
    const float* oW  = (const float*)d_in[4];
    const float* f1W = (const float*)d_in[5];
    const float* f2W = (const float*)d_in[6];
    const float* qlW = (const float*)d_in[7];
    const float* ckW = (const float*)d_in[8];
    const float* cvW = (const float*)d_in[9];
    const float* coW = (const float*)d_in[10];
    const float* cf1W= (const float*)d_in[11];
    const float* cf2W= (const float*)d_in[12];
    const float* qb  = (const float*)d_in[13];
    const float* kb  = (const float*)d_in[14];
    const float* vb  = (const float*)d_in[15];
    const float* ob  = (const float*)d_in[16];
    const float* f1b = (const float*)d_in[17];
    const float* f2b = (const float*)d_in[18];
    const float* qlb = (const float*)d_in[19];
    const float* ckb = (const float*)d_in[20];
    const float* cvb = (const float*)d_in[21];
    const float* cob = (const float*)d_in[22];
    const float* cf1b= (const float*)d_in[23];
    const float* cf2b= (const float*)d_in[24];
    const float* lnag= (const float*)d_in[25];
    const float* lnab= (const float*)d_in[26];
    const float* lnbg= (const float*)d_in[27];
    const float* lnbb= (const float*)d_in[28];
    const float* lncg= (const float*)d_in[29];
    const float* lncb= (const float*)d_in[30];
    const float* clng= (const float*)d_in[31];
    const float* clnb= (const float*)d_in[32];
    const float* cw  = (const float*)d_in[33];
    const float* flng= (const float*)d_in[34];
    const float* flnb= (const float*)d_in[35];
    const float* oW1 = (const float*)d_in[36];
    const float* ob1 = (const float*)d_in[37];
    const float* oW2 = (const float*)d_in[38];
    const float* ob2 = (const float*)d_in[39];

    float* ws = (float*)d_ws;
    float* X    = ws;
    float* Hb   = ws + 1 * SS;
    float* Qb   = ws + 2 * SS;
    float* Kb2  = ws + 3 * SS;
    float* Vb   = ws + 4 * SS;
    float* ATTb = ws + 5 * SS;
    float* SMb  = ws + 6 * SS;
    float* XQb  = ws + 7 * SS;
    float* FFb  = ws + 8 * SS;                       // Rr*SFf reuse buffer
    float* YLNb = FFb + (ll)Rr * SFf;                // [B, K]
    float* HIDb = YLNb + (ll)Bb * KK;                // [B, E]

    hipMemcpyAsync(X, x_in, (size_t)SS * sizeof(float), hipMemcpyDeviceToDevice, stream);

    const ll DD = (ll)Dd * Dd;
    const int lnBlocks = Mm * Bb * Nn;
    const int mixBlocks = (int)((SS + 255) / 256);

    for (int d = 0; d < DEPTHd; ++d) {
        const float* qW_d  = qW  + (ll)d * Mm * DD;
        const float* kW_d  = kW  + (ll)d * Mm * DD;
        const float* vW_d  = vW  + (ll)d * Mm * DD;
        const float* oW_d  = oW  + (ll)d * Mm * DD;
        const float* qlW_d = qlW + (ll)d * Mm * DD;
        const float* ckW_d = ckW + (ll)d * Mm * DD;
        const float* cvW_d = cvW + (ll)d * Mm * DD;
        const float* coW_d = coW + (ll)d * Mm * DD;
        const float* f1W_d = f1W + (ll)d * Mm * Dd * Ff;
        const float* f2W_d = f2W + (ll)d * Mm * Ff * Dd;
        const float* cf1W_d= cf1W+ (ll)d * Mm * Dd * SFf;
        const float* cf2W_d= cf2W+ (ll)d * Mm * SFf * Dd;
        const float* qb_d  = qb  + (ll)d * Mm * Dd;
        const float* kb_d  = kb  + (ll)d * Mm * Dd;
        const float* vb_d  = vb  + (ll)d * Mm * Dd;
        const float* ob_d  = ob  + (ll)d * Mm * Dd;
        const float* qlb_d = qlb + (ll)d * Mm * Dd;
        const float* ckb_d = ckb + (ll)d * Mm * Dd;
        const float* cvb_d = cvb + (ll)d * Mm * Dd;
        const float* cob_d = cob + (ll)d * Mm * Dd;
        const float* f1b_d = f1b + (ll)d * Mm * Ff;
        const float* f2b_d = f2b + (ll)d * Mm * Dd;
        const float* cf1b_d= cf1b+ (ll)d * Mm * SFf;
        const float* cf2b_d= cf2b+ (ll)d * Mm * Dd;
        const float* cw_d  = cw  + (ll)d * Mm * Mm;

        // --- self attention block ---
        ln_kernel<<<lnBlocks, 256, 0, stream>>>(X, Hb, lnag + (ll)d * Mm * Dd, lnab + (ll)d * Mm * Dd, Dd, Bb * Nn);
        launch_gemm(stream, Hb, S1, qW_d, DD, qb_d, Dd, nullptr, 0, Qb,  S1, Rr, Dd, Dd, Mm, 0);
        launch_gemm(stream, Hb, S1, kW_d, DD, kb_d, Dd, nullptr, 0, Kb2, S1, Rr, Dd, Dd, Mm, 0);
        launch_gemm(stream, Hb, S1, vW_d, DD, vb_d, Dd, nullptr, 0, Vb,  S1, Rr, Dd, Dd, Mm, 0);
        attn_kernel<<<Mm * Bb * Hh, 256, 0, stream>>>(Qb, Kb2, Vb, ATTb);
        launch_gemm(stream, ATTb, S1, oW_d, DD, ob_d, Dd, Hb, S1, X, S1, Rr, Dd, Dd, Mm, 0);

        // --- per-modality FFN ---
        ln_kernel<<<lnBlocks, 256, 0, stream>>>(X, Hb, lnbg + (ll)d * Mm * Dd, lnbb + (ll)d * Mm * Dd, Dd, Bb * Nn);
        for (int m = 0; m < Mm; ++m) {
            launch_gemm(stream, Hb + (ll)m * S1, 0, f1W_d + (ll)m * Dd * Ff, 0, f1b_d + (ll)m * Ff, 0,
                        nullptr, 0, FFb, 0, Rr, Dd, Ff, 1, 1);
            launch_gemm(stream, FFb, 0, f2W_d + (ll)m * Ff * Dd, 0, f2b_d + (ll)m * Dd, 0,
                        Hb + (ll)m * S1, 0, ATTb + (ll)m * S1, 0, Rr, Ff, Dd, 1, 0);
        }

        // --- cross-channel attention block ---
        ln_kernel<<<lnBlocks, 256, 0, stream>>>(ATTb, Hb, lncg + (ll)d * Mm * Dd, lncb + (ll)d * Mm * Dd, Dd, Bb * Nn);
        launch_gemm(stream, Hb, S1, qlW_d, DD, qlb_d, Dd, nullptr, 0, XQb, S1, Rr, Dd, Dd, Mm, 0);
        mix_kernel<<<mixBlocks, 256, 0, stream>>>(Hb, cw_d, SMb);
        launch_gemm(stream, SMb, S1, ckW_d, DD, ckb_d, Dd, nullptr, 0, Kb2, S1, Rr, Dd, Dd, Mm, 0);
        launch_gemm(stream, SMb, S1, cvW_d, DD, cvb_d, Dd, nullptr, 0, Vb,  S1, Rr, Dd, Dd, Mm, 0);
        attn_kernel<<<Mm * Bb * Hh, 256, 0, stream>>>(XQb, Kb2, Vb, Qb);
        launch_gemm(stream, Qb, S1, coW_d, DD, cob_d, Dd, XQb, S1, ATTb, S1, Rr, Dd, Dd, Mm, 0);

        // --- cross FFN ---
        ln_kernel<<<lnBlocks, 256, 0, stream>>>(ATTb, Hb, clng + (ll)d * Mm * Dd, clnb + (ll)d * Mm * Dd, Dd, Bb * Nn);
        for (int m = 0; m < Mm; ++m) {
            launch_gemm(stream, Hb + (ll)m * S1, 0, cf1W_d + (ll)m * Dd * SFf, 0, cf1b_d + (ll)m * SFf, 0,
                        nullptr, 0, FFb, 0, Rr, Dd, SFf, 1, 1);
            launch_gemm(stream, FFb, 0, cf2W_d + (ll)m * SFf * Dd, 0, cf2b_d + (ll)m * Dd, 0,
                        Hb + (ll)m * S1, 0, X + (ll)m * S1, 0, Rr, SFf, Dd, 1, 0);
        }
    }

    // --- output head ---
    final_ln_kernel<<<Bb, 256, 0, stream>>>(X, YLNb, flng, flnb);
    launch_gemm(stream, YLNb, 0, oW1, 0, ob1, 0, nullptr, 0, HIDb, 0, Bb, KK, Ee, 1, 1);
    launch_gemm(stream, HIDb, 0, oW2, 0, ob2, 0, nullptr, 0, (float*)d_out, 0, Bb, Ee, OUTo, 1, 0);
}
